// GATNet_7713761263899
// MI455X (gfx1250) — compile-verified, matrix operand order checked by reading, not run on hardware
//
#include <hip/hip_runtime.h>

// ---------------------------------------------------------------------------
// GAT (2-layer, PyG-style) for MI455X / gfx1250.
//  - Dense projections: bf16 WMMA (v_wmma_f32_16x16x32_bf16), f32 accumulate,
//    K fully unrolled via template (4 static WMMAs per tile).
//  - Edge softmax: 3-pass (atomic max / exp-sum / weighted scatter); the
//    [N,128] accumulator (51MB) is L2-resident (192MB L2), so the ~200M f32
//    scatter atomics stay on-chip; gathers stream at HBM speed.
// ---------------------------------------------------------------------------

typedef __attribute__((ext_vector_type(16))) __bf16 v16bf;
typedef __attribute__((ext_vector_type(8)))  float  v8f;

__device__ __forceinline__ float lrelu(float x) { return x > 0.f ? x : 0.2f * x; }

__device__ __forceinline__ void atomicMaxF(float* addr, float val) {
    // valid for the full float range incl. negatives / -inf init
    if (val >= 0.f) atomicMax((int*)addr, __float_as_int(val));
    else            atomicMin((unsigned int*)addr, __float_as_uint(val));
}

// ---------------------------------------------------------------------------
// Y[n,M] = X[n,K] @ W[K,M], bf16 operands (native cvt in-register), f32 accum.
// One wave = one 16x16 output tile; K stepped by 32 per WMMA, fully unrolled.
// ---------------------------------------------------------------------------
template <int K, int M>
__global__ void gemm_bf16_wmma(const float* __restrict__ X,
                               const float* __restrict__ W,
                               float* __restrict__ Y,
                               int n) {
    const int lane = threadIdx.x & 31;
    const int wib  = threadIdx.x >> 5;
    const int wpb  = blockDim.x >> 5;
    const int wid  = blockIdx.x * wpb + wib;           // wave-uniform
    constexpr int colTiles = M >> 4;
    const int rowTiles = (n + 15) >> 4;
    if (wid >= rowTiles * colTiles) return;            // whole-wave exit
    const int r0 = (wid / colTiles) << 4;
    const int n0 = (wid % colTiles) << 4;

    const int m  = lane & 15;      // A row within tile / B,C,D column
    const int kh = lane >> 4;      // K-half selector (ISA 16-bit A/B layout)

    int row = r0 + m;
    if (row >= n) row = n - 1;     // clamp loads only; EXEC stays all-ones
    const float* xrow = X + (size_t)row * K;

    v8f acc = {};
#pragma unroll
    for (int k0 = 0; k0 < K; k0 += 32) {
        const float* xr = xrow + k0;
        v16bf a, b;
#pragma unroll
        for (int j = 0; j < 8; ++j) {
            a[j]     = (__bf16)xr[8 * kh + j];          // K = 8*kh + j
            a[8 + j] = (__bf16)xr[16 + 8 * kh + j];     // K = 16 + 8*kh + j
        }
#pragma unroll
        for (int j = 0; j < 8; ++j) {
            b[j]     = (__bf16)W[(k0 + 8 * kh + j) * M + n0 + m];
            b[8 + j] = (__bf16)W[(k0 + 16 + 8 * kh + j) * M + n0 + m];
        }
        acc = __builtin_amdgcn_wmma_f32_16x16x32_bf16(
                  false, a, false, b, (short)0, acc, false, false);
    }

    // C/D layout: VGPR v -> row r0 + v + 8*kh, column n0 + m
    float* yp = Y + (size_t)(r0 + 8 * kh) * M + n0 + m;
    if (r0 + 16 <= n) {                                // wave-uniform fast path
#pragma unroll
        for (int v = 0; v < 8; ++v) yp[(size_t)v * M] = acc[v];
    } else {
#pragma unroll
        for (int v = 0; v < 8; ++v)
            if (r0 + 8 * kh + v < n) yp[(size_t)v * M] = acc[v];
    }
}

// ---------------------------------------------------------------------------
// Per-(node,head) attention logits: as = <h[n,h,:], a_src[h,:]>, same for ad.
// ---------------------------------------------------------------------------
__global__ void alpha_kernel(const float* __restrict__ h,
                             const float* __restrict__ a_src,
                             const float* __restrict__ a_dst,
                             float* __restrict__ as_out, float* __restrict__ ad_out,
                             int n, int H, int C) {
    int tid = blockIdx.x * blockDim.x + threadIdx.x;
    if (tid >= n * H) return;
    int node = tid / H, hh = tid - node * H;
    const float* hp = h + (size_t)node * H * C + (size_t)hh * C;
    float s = 0.f, d = 0.f;
    for (int c = 0; c < C; ++c) {
        float v = hp[c];
        s += v * a_src[hh * C + c];
        d += v * a_dst[hh * C + c];
    }
    as_out[tid] = s;
    ad_out[tid] = d;
}

// ---------------------------------------------------------------------------
// Edge passes (selfloop=1 => src=dst=edge index, covers add_self_loops)
// ---------------------------------------------------------------------------
__global__ void edge_max_kernel(const int* __restrict__ src, const int* __restrict__ dst,
                                int count, int selfloop, int H,
                                const float* __restrict__ as, const float* __restrict__ ad,
                                float* __restrict__ mx) {
    int tid = blockIdx.x * blockDim.x + threadIdx.x;
    if (tid >= count * H) return;
    int e = tid / H, h = tid - e * H;
    int s = selfloop ? e : src[e];
    int d = selfloop ? e : dst[e];
    float ev = lrelu(as[s * H + h] + ad[d * H + h]);
    atomicMaxF(&mx[d * H + h], ev);
}

__global__ void edge_expsum_kernel(const int* __restrict__ src, const int* __restrict__ dst,
                                   int count, int selfloop, int H,
                                   const float* __restrict__ as, const float* __restrict__ ad,
                                   const float* __restrict__ mx, float* __restrict__ sum) {
    int tid = blockIdx.x * blockDim.x + threadIdx.x;
    if (tid >= count * H) return;
    int e = tid / H, h = tid - e * H;
    int s = selfloop ? e : src[e];
    int d = selfloop ? e : dst[e];
    float ev = lrelu(as[s * H + h] + ad[d * H + h]);
    atomicAdd(&sum[d * H + h], __expf(ev - mx[d * H + h]));
}

// C is a multiple of 4; feature rows are 64B-aligned in the workspace.
__global__ void edge_agg_kernel(const int* __restrict__ src, const int* __restrict__ dst,
                                int count, int selfloop, int H, int C,
                                const float* __restrict__ as, const float* __restrict__ ad,
                                const float* __restrict__ mx, const float* __restrict__ sum,
                                const float* __restrict__ feat, float* __restrict__ agg) {
    int tid = blockIdx.x * blockDim.x + threadIdx.x;
    if (tid >= count * H) return;
    int e = tid / H, h = tid - e * H;
    int s = selfloop ? e : src[e];
    int d = selfloop ? e : dst[e];
    const float4* fs = (const float4*)(feat + (size_t)s * H * C + (size_t)h * C);
    __builtin_prefetch(fs, 0, 0);   // global_prefetch_b8: cover gather latency
    float ev    = lrelu(as[s * H + h] + ad[d * H + h]);
    float alpha = __expf(ev - mx[d * H + h]) / (sum[d * H + h] + 1e-16f);
    float* od = agg + (size_t)d * H * C + (size_t)h * C;
    for (int c4 = 0; c4 < (C >> 2); ++c4) {
        float4 f = fs[c4];
        atomicAdd(&od[4 * c4 + 0], alpha * f.x);
        atomicAdd(&od[4 * c4 + 1], alpha * f.y);
        atomicAdd(&od[4 * c4 + 2], alpha * f.z);
        atomicAdd(&od[4 * c4 + 3], alpha * f.w);
    }
}

// ---------------------------------------------------------------------------
// agg += bias ; ELU  (in place) — layer-1 activation
// ---------------------------------------------------------------------------
__global__ void bias_elu_kernel(float* __restrict__ agg, const float* __restrict__ bias,
                                int n, int Dm) {
    int tid = blockIdx.x * blockDim.x + threadIdx.x;
    if (tid >= n * Dm) return;
    int c = tid & (Dm - 1);        // Dm is a power of two (128)
    float v = agg[tid] + bias[c];
    agg[tid] = v > 0.f ? v : __expf(v) - 1.f;
}

// ---------------------------------------------------------------------------
// out[n] = elu(agg2[n,:] + b2) . Wh + bh
// ---------------------------------------------------------------------------
__global__ void head_kernel(const float* __restrict__ agg2, const float* __restrict__ b2,
                            const float* __restrict__ Wh, const float* __restrict__ bh,
                            float* __restrict__ out, int n, int C) {
    int node = blockIdx.x * blockDim.x + threadIdx.x;
    if (node >= n) return;
    const float* a = agg2 + (size_t)node * C;
    float acc = 0.f;
    for (int c = 0; c < C; ++c) {
        float v = a[c] + b2[c];
        v = v > 0.f ? v : __expf(v) - 1.f;
        acc += v * Wh[c];
    }
    out[node] = acc + bh[0];
}

__global__ void fill_kernel(float* __restrict__ p, float v, size_t cnt) {
    size_t i = (size_t)blockIdx.x * blockDim.x + threadIdx.x;
    if (i < cnt) p[i] = v;
}

// ---------------------------------------------------------------------------
extern "C" void kernel_launch(void* const* d_in, const int* in_sizes, int n_in,
                              void* d_out, int out_size, void* d_ws, size_t ws_size,
                              hipStream_t stream) {
    const float* x   = (const float*)d_in[0];
    const int*   ei  = (const int*)  d_in[1];
    const float* W1  = (const float*)d_in[2];
    const float* a1s = (const float*)d_in[3];
    const float* a1d = (const float*)d_in[4];
    const float* b1  = (const float*)d_in[5];
    const float* W2  = (const float*)d_in[6];
    const float* a2s = (const float*)d_in[7];
    const float* a2d = (const float*)d_in[8];
    const float* b2  = (const float*)d_in[9];
    const float* Wh  = (const float*)d_in[10];
    const float* bh  = (const float*)d_in[11];
    float* out = (float*)d_out;

    const int Dm = 128, H1 = 8, C1 = 16, C2 = 32;
    const int n = in_sizes[0] / Dm;
    const int E = in_sizes[1] / 2;
    const int* src = ei;        // edge_index[0]
    const int* dst = ei + E;    // edge_index[1]

    // workspace carve-out (floats)
    float* ws = (float*)d_ws;
    float* h1   = ws; ws += (size_t)n * 128;
    float* as1  = ws; ws += (size_t)n * 8;
    float* ad1  = ws; ws += (size_t)n * 8;
    float* m1   = ws; ws += (size_t)n * 8;
    float* s1   = ws; ws += (size_t)n * 8;
    float* agg1 = ws; ws += (size_t)n * 128;   // becomes layer-2 input after bias+elu
    float* h2   = ws; ws += (size_t)n * 32;
    float* as2  = ws; ws += (size_t)n;
    float* ad2  = ws; ws += (size_t)n;
    float* m2   = ws; ws += (size_t)n;
    float* s2   = ws; ws += (size_t)n;
    float* agg2 = ws; ws += (size_t)n * 32;

    const int TB = 256;
    auto blocks = [](long long work, int tb) { return (unsigned)((work + tb - 1) / tb); };

    // ---- init accumulators (deterministic each call) ----
    fill_kernel<<<blocks((long long)n * 8, TB), TB, 0, stream>>>(m1, -INFINITY, (size_t)n * 8);
    fill_kernel<<<blocks((long long)n * 8, TB), TB, 0, stream>>>(s1, 0.f, (size_t)n * 8);
    fill_kernel<<<blocks((long long)n * 128, TB), TB, 0, stream>>>(agg1, 0.f, (size_t)n * 128);
    fill_kernel<<<blocks(n, TB), TB, 0, stream>>>(m2, -INFINITY, (size_t)n);
    fill_kernel<<<blocks(n, TB), TB, 0, stream>>>(s2, 0.f, (size_t)n);
    fill_kernel<<<blocks((long long)n * 32, TB), TB, 0, stream>>>(agg2, 0.f, (size_t)n * 32);

    // ---- layer 1: projection (WMMA) + logits ----
    {
        long long tiles = (long long)((n + 15) / 16) * (128 / 16);
        gemm_bf16_wmma<128, 128><<<blocks(tiles, 8), TB, 0, stream>>>(x, W1, h1, n);
        alpha_kernel<<<blocks((long long)n * H1, TB), TB, 0, stream>>>(h1, a1s, a1d, as1, ad1, n, H1, C1);
    }
    // ---- layer 1: segment softmax + aggregate (edges, then self loops) ----
    edge_max_kernel   <<<blocks((long long)E * H1, TB), TB, 0, stream>>>(src, dst, E, 0, H1, as1, ad1, m1);
    edge_max_kernel   <<<blocks((long long)n * H1, TB), TB, 0, stream>>>(src, dst, n, 1, H1, as1, ad1, m1);
    edge_expsum_kernel<<<blocks((long long)E * H1, TB), TB, 0, stream>>>(src, dst, E, 0, H1, as1, ad1, m1, s1);
    edge_expsum_kernel<<<blocks((long long)n * H1, TB), TB, 0, stream>>>(src, dst, n, 1, H1, as1, ad1, m1, s1);
    edge_agg_kernel   <<<blocks((long long)E * H1, TB), TB, 0, stream>>>(src, dst, E, 0, H1, C1, as1, ad1, m1, s1, h1, agg1);
    edge_agg_kernel   <<<blocks((long long)n * H1, TB), TB, 0, stream>>>(src, dst, n, 1, H1, C1, as1, ad1, m1, s1, h1, agg1);
    // bias + ELU -> layer-2 node features (in place)
    bias_elu_kernel<<<blocks((long long)n * 128, TB), TB, 0, stream>>>(agg1, b1, n, 128);

    // ---- layer 2: projection (WMMA) + logits ----
    {
        long long tiles = (long long)((n + 15) / 16) * (32 / 16);
        gemm_bf16_wmma<128, 32><<<blocks(tiles, 8), TB, 0, stream>>>(agg1, W2, h2, n);
        alpha_kernel<<<blocks(n, TB), TB, 0, stream>>>(h2, a2s, a2d, as2, ad2, n, 1, C2);
    }
    // ---- layer 2: segment softmax + aggregate ----
    edge_max_kernel   <<<blocks(E, TB), TB, 0, stream>>>(src, dst, E, 0, 1, as2, ad2, m2);
    edge_max_kernel   <<<blocks(n, TB), TB, 0, stream>>>(src, dst, n, 1, 1, as2, ad2, m2);
    edge_expsum_kernel<<<blocks(E, TB), TB, 0, stream>>>(src, dst, E, 0, 1, as2, ad2, m2, s2);
    edge_expsum_kernel<<<blocks(n, TB), TB, 0, stream>>>(src, dst, n, 1, 1, as2, ad2, m2, s2);
    edge_agg_kernel   <<<blocks(E, TB), TB, 0, stream>>>(src, dst, E, 0, 1, C2, as2, ad2, m2, s2, h2, agg2);
    edge_agg_kernel   <<<blocks(n, TB), TB, 0, stream>>>(src, dst, n, 1, 1, C2, as2, ad2, m2, s2, h2, agg2);

    // ---- bias + ELU + ordinal head ----
    head_kernel<<<blocks(n, TB), TB, 0, stream>>>(agg2, b2, Wh, bh, out, n, C2);
}
